// MPSGNN_11570641895845
// MI455X (gfx1250) — matrix operand order repaired and checked by reading, not compiled, not run on hardware
//
#include <hip/hip_runtime.h>
#include <hip/hip_bf16.h>

// ---------------------------------------------------------------------------
// Problem constants (from reference)
// ---------------------------------------------------------------------------
#define NA 40000
#define NB 80000
#define CC 256
#define MM 4
#define EE 250000
#define CC2 (CC * CC)

typedef __attribute__((ext_vector_type(16))) __bf16 v16bf;
typedef __attribute__((ext_vector_type(8)))  float  v8f;

// ---------------------------------------------------------------------------
// Conversion kernels
// ---------------------------------------------------------------------------
__global__ __launch_bounds__(256) void mps_cvt_bf16(const float* __restrict__ in,
                                                    __bf16* __restrict__ out, long n) {
  long t = ((long)blockIdx.x * blockDim.x + threadIdx.x) * 4;
  if (t + 3 < n) {
    float4 v = *reinterpret_cast<const float4*>(in + t);
    out[t + 0] = (__bf16)v.x; out[t + 1] = (__bf16)v.y;
    out[t + 2] = (__bf16)v.z; out[t + 3] = (__bf16)v.w;
  } else {
    for (long i = t; i < n; ++i) out[i] = (__bf16)in[i];
  }
}

// in: [K][N] f32 row-major -> out: [N][K] bf16 row-major (B^T layout for GEMM)
__global__ __launch_bounds__(256) void mps_cvt_transpose_bf16(const float* __restrict__ in,
                                                              __bf16* __restrict__ out,
                                                              int K, int N) {
  unsigned t = blockIdx.x * blockDim.x + threadIdx.x;
  if (t >= (unsigned)(K * N)) return;
  unsigned n = t / (unsigned)K, k = t % (unsigned)K;
  out[t] = (__bf16)in[(size_t)k * N + n];
}

// ---------------------------------------------------------------------------
// Edge scatter-mean (segment sum via f32 atomics, 4 channels/lane)
// ---------------------------------------------------------------------------
__global__ __launch_bounds__(256) void mps_scatter_f32(const float* __restrict__ xsrc,
                                                       const int* __restrict__ src,
                                                       const int* __restrict__ dst,
                                                       float* __restrict__ agg,
                                                       float* __restrict__ cnt, int E) {
  unsigned t = blockIdx.x * blockDim.x + threadIdx.x;
  unsigned e = t >> 6, q = t & 63;
  if (e >= (unsigned)E) return;
  int s = src[e], d = dst[e];
  float4 v = *reinterpret_cast<const float4*>(xsrc + (size_t)s * CC + q * 4);
  float* ap = agg + (size_t)d * CC + q * 4;
  atomicAdd(ap + 0, v.x); atomicAdd(ap + 1, v.y);
  atomicAdd(ap + 2, v.z); atomicAdd(ap + 3, v.w);
  if (q == 0) atomicAdd(cnt + d, 1.0f);
}

__global__ __launch_bounds__(256) void mps_scatter_bf16(const __bf16* __restrict__ xsrc,
                                                        const int* __restrict__ src,
                                                        const int* __restrict__ dst,
                                                        float* __restrict__ agg,
                                                        float* __restrict__ cnt, int E) {
  unsigned t = blockIdx.x * blockDim.x + threadIdx.x;
  unsigned e = t >> 6, q = t & 63;
  if (e >= (unsigned)E) return;
  int s = src[e], d = dst[e];
  const __bf16* sp = xsrc + (size_t)s * CC + q * 4;
  float* ap = agg + (size_t)d * CC + q * 4;
  atomicAdd(ap + 0, (float)sp[0]); atomicAdd(ap + 1, (float)sp[1]);
  atomicAdd(ap + 2, (float)sp[2]); atomicAdd(ap + 3, (float)sp[3]);
  if (q == 0) atomicAdd(cnt + d, 1.0f);
}

// mean = agg / max(cnt,1) -> bf16
__global__ __launch_bounds__(256) void mps_mean_bf16(const float* __restrict__ agg,
                                                     const float* __restrict__ cnt,
                                                     __bf16* __restrict__ out, int rows) {
  unsigned t = blockIdx.x * blockDim.x + threadIdx.x;
  unsigned r = t >> 6, q = t & 63;
  if (r >= (unsigned)rows) return;
  float inv = 1.0f / fmaxf(cnt[r], 1.0f);
  float4 v = *reinterpret_cast<const float4*>(agg + (size_t)r * CC + q * 4);
  __bf16* op = out + (size_t)r * CC + q * 4;
  op[0] = (__bf16)(v.x * inv); op[1] = (__bf16)(v.y * inv);
  op[2] = (__bf16)(v.z * inv); op[3] = (__bf16)(v.w * inv);
}

// ---------------------------------------------------------------------------
// WMMA bf16 GEMM:  out = scale * ( relu?( A1*B1t^T [+ A2*B2t^T] + bias ) )
//   A*: [rows][K] bf16 row-major.  B*t: [N][K] bf16 (pre-transposed weights).
//   Block tile 128x128, 8 wave32 waves, each wave 32x64 (2x4 16x16 frags).
//   K-tiles staged via GLOBAL_LOAD_ASYNC_TO_LDS_B128 with double buffering;
//   consumer waits with s_wait_asynccnt, then WMMA from LDS (ds_load_b128).
// ---------------------------------------------------------------------------
#define LDT 40                  // padded tile row stride in bf16 (80B -> no bank conflict)
#define TSZ (128 * LDT)         // elements per tile

// Issue one row-half (32B) of a tile as two async b128 copies.
// LDS destination is l0 + offset; global source is sbase + goff + offset
__device__ __forceinline__ void mps_stage_tile(__bf16* tile, uint32_t telem,
                                               uint32_t goff, const __bf16* sbase) {
  uint32_t l0 = (uint32_t)(uintptr_t)(tile + telem);
  asm volatile("global_load_async_to_lds_b128 %0, %1, %2"
               :: "v"(l0), "v"(goff), "s"(sbase) : "memory");
  asm volatile("global_load_async_to_lds_b128 %0, %1, %2 offset:16"
               :: "v"(l0), "v"(goff), "s"(sbase) : "memory");
}

__device__ __forceinline__ v16bf mps_ldsA_frag(const __bf16* t, int lane, int row0) {
  // 16-bit A 16x32 layout: lanes 0-15 -> K {0..7,16..23}; lanes 16-31 -> K {8..15,24..31}
  int r  = row0 + (lane & 15);
  int kb = (lane >> 4) << 3;
  const __bf16* p = t + r * LDT + kb;
  union { uint4 u[2]; v16bf v; } x;
  x.u[0] = *reinterpret_cast<const uint4*>(p);
  x.u[1] = *reinterpret_cast<const uint4*>(p + 16);
  return x.v;
}

__device__ __forceinline__ v16bf mps_ldsB_frag(const __bf16* t, int lane, int col0) {
  // 16-bit B 32x16 layout: lanes 0-15 -> K {0..7,16..23}; lanes 16-31 -> K {8..15,24..31}
  int c  = col0 + (lane & 15);
  int kh = (lane >> 4) << 3;
  const __bf16* p = t + c * LDT + kh;
  union { uint4 u[2]; v16bf v; } x;
  x.u[0] = *reinterpret_cast<const uint4*>(p);
  x.u[1] = *reinterpret_cast<const uint4*>(p + 16);
  return x.v;
}

__global__ __launch_bounds__(256) void mps_wmma_gemm(
    const __bf16* __restrict__ A1, const __bf16* __restrict__ B1t,
    const __bf16* __restrict__ A2, const __bf16* __restrict__ B2t,
    const float* __restrict__ bias, const float* __restrict__ scale_ptr,
    float* __restrict__ outF, __bf16* __restrict__ outB,
    int rows, int N, int K, int out_stride, int relu) {
  extern __shared__ __bf16 mps_smem[];
  // Tile index layout (all pointers computed at runtime; NEVER store the
  // dynamic-LDS symbol into const-initialized arrays -> lld relocation error):
  //   buf b: A1 -> (2b+0)*TSZ, B1 -> (2b+1)*TSZ, A2 -> (4+2b)*TSZ, B2 -> (5+2b)*TSZ

  const int rowBase = blockIdx.x * 128;
  const int colBase = blockIdx.y * 128;
  const int lane = threadIdx.x & 31;
  const int wave = threadIdx.x >> 5;
  const int wm = wave >> 1;   // 0..3 -> +32 rows
  const int wn = wave & 1;    // 0..1 -> +64 cols
  const bool has2 = (A2 != nullptr);

  v8f acc[2][4];
#pragma unroll
  for (int i = 0; i < 2; ++i)
#pragma unroll
    for (int j = 0; j < 4; ++j) acc[i][j] = (v8f){};

  // Per-thread staging slot: row lr, K-half `half` (16 bf16 = 32B per thread per tile).
  const int lr   = threadIdx.x >> 1;   // 0..127
  const int half = threadIdx.x & 1;
  // Tail A rows clamped into range: D-row depends only on same A-row, and tail
  // rows are masked in the epilogue, so clamped (garbage) rows are harmless.
  const int grA = (rowBase + lr < rows) ? (rowBase + lr) : (rows - 1);
  const uint32_t aoff  = (uint32_t)((size_t)grA * (size_t)K * 2u) + (uint32_t)(half * 32);
  const uint32_t boff  = (uint32_t)((size_t)(colBase + lr) * (size_t)K * 2u) + (uint32_t)(half * 32);
  const uint32_t telem = (uint32_t)(lr * LDT + half * 16);

  // Prologue: stage K-tile 0 into buffer 0.
  {
    int b0 = 0;  // runtime-ish index keeps pointer math in the instruction stream
    mps_stage_tile(mps_smem + (2 * b0 + 0) * TSZ, telem, aoff, A1);
    mps_stage_tile(mps_smem + (2 * b0 + 1) * TSZ, telem, boff, B1t);
    if (has2) {
      mps_stage_tile(mps_smem + (4 + 2 * b0) * TSZ, telem, aoff, A2);
      mps_stage_tile(mps_smem + (5 + 2 * b0) * TSZ, telem, boff, B2t);
    }
  }

  int cb = 0;
  for (int k0 = 0; k0 < K; k0 += 32) {
    const bool more = (k0 + 32 < K);
    if (more) {  // stage next K-tile into the other buffer (overlaps WMMA below)
      const int nb = cb ^ 1;
      mps_stage_tile(mps_smem + (2 * nb + 0) * TSZ, telem, aoff, A1 + k0 + 32);
      mps_stage_tile(mps_smem + (2 * nb + 1) * TSZ, telem, boff, B1t + k0 + 32);
      if (has2) {
        mps_stage_tile(mps_smem + (4 + 2 * nb) * TSZ, telem, aoff, A2 + k0 + 32);
        mps_stage_tile(mps_smem + (5 + 2 * nb) * TSZ, telem, boff, B2t + k0 + 32);
      }
      // wait until only the just-issued stage is outstanding
      if (has2) asm volatile("s_wait_asynccnt 0x8" ::: "memory");
      else      asm volatile("s_wait_asynccnt 0x4" ::: "memory");
    } else {
      asm volatile("s_wait_asynccnt 0x0" ::: "memory");
    }
    __syncthreads();

    // ---- WMMA on 32x64 wave tile from buffer cb ----
    {
      const __bf16* ta1 = mps_smem + (2 * cb + 0) * TSZ;
      const __bf16* tb1 = mps_smem + (2 * cb + 1) * TSZ;
      v16bf a0 = mps_ldsA_frag(ta1, lane, wm * 32);
      v16bf a1 = mps_ldsA_frag(ta1, lane, wm * 32 + 16);
#pragma unroll
      for (int j = 0; j < 4; ++j) {
        v16bf b = mps_ldsB_frag(tb1, lane, wn * 64 + j * 16);
        acc[0][j] = __builtin_amdgcn_wmma_f32_16x16x32_bf16(false, a0, false, b,
                                                            (short)0, acc[0][j], false, false);
        acc[1][j] = __builtin_amdgcn_wmma_f32_16x16x32_bf16(false, a1, false, b,
                                                            (short)0, acc[1][j], false, false);
      }
      if (has2) {
        const __bf16* ta2 = mps_smem + (4 + 2 * cb) * TSZ;
        const __bf16* tb2 = mps_smem + (5 + 2 * cb) * TSZ;
        v16bf c0 = mps_ldsA_frag(ta2, lane, wm * 32);
        v16bf c1 = mps_ldsA_frag(ta2, lane, wm * 32 + 16);
#pragma unroll
        for (int j = 0; j < 4; ++j) {
          v16bf b = mps_ldsB_frag(tb2, lane, wn * 64 + j * 16);
          acc[0][j] = __builtin_amdgcn_wmma_f32_16x16x32_bf16(false, c0, false, b,
                                                              (short)0, acc[0][j], false, false);
          acc[1][j] = __builtin_amdgcn_wmma_f32_16x16x32_bf16(false, c1, false, b,
                                                              (short)0, acc[1][j], false, false);
        }
      }
    }
    __syncthreads();   // buffer cb free for stage k0+64
    cb ^= 1;
  }

  // ---- epilogue ----
  const float scale = scale_ptr ? scale_ptr[0] : 1.0f;
#pragma unroll
  for (int j = 0; j < 4; ++j) {
    int c = colBase + wn * 64 + j * 16 + (lane & 15);
    float bv = bias ? bias[c] : 0.0f;
#pragma unroll
    for (int mi = 0; mi < 2; ++mi) {
      union { v8f v; float f[8]; } u;
      u.v = acc[mi][j];
      int rb = rowBase + wm * 32 + mi * 16 + ((lane >> 4) << 3);
#pragma unroll
      for (int i = 0; i < 8; ++i) {
        int r = rb + i;
        if (r < rows) {
          float val = u.f[i] + bv;
          if (relu) val = fmaxf(val, 0.0f);
          val *= scale;
          if (outF) outF[(size_t)r * out_stride + c] = val;
          if (outB) outB[(size_t)r * out_stride + c] = (__bf16)val;
        }
      }
    }
  }
}

// ---------------------------------------------------------------------------
// Multihead attention over metapath axis (M=4, 4 heads, d=64).
// One wave per (node, head): lane owns 2 channels, 4x4 scores via shfl reduce.
// ---------------------------------------------------------------------------
__global__ __launch_bounds__(128) void mps_attention(const __bf16* __restrict__ qkv,
                                                     __bf16* __restrict__ o, int nA) {
  int n = blockIdx.x;
  if (n >= nA) return;
  int h = threadIdx.x >> 5;
  int lane = threadIdx.x & 31;
  const __bf16* base = qkv + (size_t)n * (MM * 3 * CC);
  int c0 = h * 64 + lane * 2;

  float q[MM][2], k[MM][2], v[MM][2];
#pragma unroll
  for (int m = 0; m < MM; ++m) {
    const __bf16* row = base + m * (3 * CC);
    q[m][0] = (float)row[c0];            q[m][1] = (float)row[c0 + 1];
    k[m][0] = (float)row[CC + c0];       k[m][1] = (float)row[CC + c0 + 1];
    v[m][0] = (float)row[2 * CC + c0];   v[m][1] = (float)row[2 * CC + c0 + 1];
  }
  float s[MM][MM];
#pragma unroll
  for (int mq = 0; mq < MM; ++mq)
#pragma unroll
    for (int mk = 0; mk < MM; ++mk) {
      float p = q[mq][0] * k[mk][0] + q[mq][1] * k[mk][1];
#pragma unroll
      for (int off = 1; off < 32; off <<= 1) p += __shfl_xor(p, off, 32);
      s[mq][mk] = p * 0.125f;  // 1/sqrt(64)
    }
#pragma unroll
  for (int mq = 0; mq < MM; ++mq) {
    float mx = fmaxf(fmaxf(s[mq][0], s[mq][1]), fmaxf(s[mq][2], s[mq][3]));
    float e0 = __expf(s[mq][0] - mx), e1 = __expf(s[mq][1] - mx);
    float e2 = __expf(s[mq][2] - mx), e3 = __expf(s[mq][3] - mx);
    float inv = 1.0f / (e0 + e1 + e2 + e3);
    float o0 = (e0 * v[0][0] + e1 * v[1][0] + e2 * v[2][0] + e3 * v[3][0]) * inv;
    float o1 = (e0 * v[0][1] + e1 * v[1][1] + e2 * v[2][1] + e3 * v[3][1]) * inv;
    __bf16* op = o + (size_t)n * (MM * CC) + mq * CC + c0;
    op[0] = (__bf16)o0; op[1] = (__bf16)o1;
  }
}

// pooled[n][c] = mean_m attn_o[n][m][c] -> bf16
__global__ __launch_bounds__(256) void mps_pool(const float* __restrict__ ao,
                                                __bf16* __restrict__ pooled, int nA) {
  unsigned t = blockIdx.x * blockDim.x + threadIdx.x;
  if (t >= (unsigned)nA * CC) return;
  unsigned n = t >> 8, c = t & 255;
  const float* p = ao + (size_t)n * (MM * CC) + c;
  float s = 0.25f * (p[0] + p[CC] + p[2 * CC] + p[3 * CC]);
  pooled[t] = (__bf16)s;
}

// out[n] = dot(h[n,:], w2) + b2   (one wave per node, shuffle reduce)
__global__ __launch_bounds__(256) void mps_final(const __bf16* __restrict__ h,
                                                 const float* __restrict__ w2,
                                                 const float* __restrict__ b2,
                                                 float* __restrict__ out, int nA) {
  int n = blockIdx.x * 8 + (threadIdx.x >> 5);
  int lane = threadIdx.x & 31;
  if (n >= nA) return;
  const __bf16* row = h + (size_t)n * CC;
  float acc = 0.0f;
#pragma unroll
  for (int j = 0; j < 8; ++j) {
    int c = lane + j * 32;
    acc += (float)row[c] * w2[c];
  }
#pragma unroll
  for (int off = 1; off < 32; off <<= 1) acc += __shfl_xor(acc, off, 32);
  if (lane == 0) out[n] = acc + b2[0];
}

// ---------------------------------------------------------------------------
// Host orchestration
// ---------------------------------------------------------------------------
extern "C" void kernel_launch(void* const* d_in, const int* in_sizes, int n_in,
                              void* d_out, int out_size, void* d_ws, size_t ws_size,
                              hipStream_t stream) {
  const float* x_a        = (const float*)d_in[0];
  const float* x_b        = (const float*)d_in[1];
  const int*   edges_ab   = (const int*)d_in[2];
  const int*   edges_ba   = (const int*)d_in[3];
  const float* wl1        = (const float*)d_in[4];
  const float* wr1        = (const float*)d_in[5];
  const float* b1         = (const float*)d_in[6];
  const float* wl2        = (const float*)d_in[7];
  const float* wr2        = (const float*)d_in[8];
  const float* b2         = (const float*)d_in[9];
  const float* outw       = (const float*)d_in[10];
  const float* outb       = (const float*)d_in[11];
  const float* mp_w       = (const float*)d_in[12];
  const float* in_proj_w  = (const float*)d_in[13];
  const float* in_proj_b  = (const float*)d_in[14];
  const float* attn_out_w = (const float*)d_in[15];
  const float* attn_out_b = (const float*)d_in[16];
  const float* mlp1_w     = (const float*)d_in[17];
  const float* mlp1_b     = (const float*)d_in[18];
  const float* mlp2_w     = (const float*)d_in[19];
  const float* mlp2_b     = (const float*)d_in[20];
  float* out = (float*)d_out;

  const unsigned SMEM2 = 8u * TSZ * 2u;  // dual-pair GEMM: 80KB dynamic LDS
  const unsigned SMEM1 = 4u * TSZ * 2u;  // single-pair GEMM: 40KB

  // ---- workspace carve (stream-serialized, so aliasing below is safe) ----
  char* p = (char*)d_ws;
  auto carve = [&](size_t bytes) -> char* {
    char* r = p; p += (bytes + 255) & ~(size_t)255; return r;
  };
  __bf16* xa_bf   = (__bf16*)carve((size_t)NA * CC * 2);
  __bf16* xb_bf   = (__bf16*)carve((size_t)NB * CC * 2);
  float*  agg     = (float*) carve((size_t)NB * CC * 4);
  float*  cnt     = (float*) carve((size_t)NB * 4);
  __bf16* meanbf  = (__bf16*)carve((size_t)NB * CC * 2);
  __bf16* h_bf    = (__bf16*)carve((size_t)NB * CC * 2);   // hb, then ha
  __bf16* xstack  = (__bf16*)carve((size_t)NA * MM * CC * 2); // embs; later o
  char*   big     =          carve((size_t)NA * MM * 3 * CC * 2); // qkv; later attn_f32
  __bf16* qkv     = (__bf16*)big;
  float*  attn_f  = (float*)big;   // alias: qkv dead once attention done
  __bf16* pooled  = (__bf16*)carve((size_t)NA * CC * 2);
  __bf16* hmlp    = (__bf16*)carve((size_t)NA * CC * 2);
  __bf16* wl1t    = (__bf16*)carve((size_t)MM * CC2 * 2);
  __bf16* wr1t    = (__bf16*)carve((size_t)MM * CC2 * 2);
  __bf16* wl2t    = (__bf16*)carve((size_t)MM * CC2 * 2);
  __bf16* wr2t    = (__bf16*)carve((size_t)MM * CC2 * 2);
  __bf16* outwt   = (__bf16*)carve((size_t)MM * CC2 * 2);
  __bf16* mlp1t   = (__bf16*)carve((size_t)CC2 * 2);
  __bf16* inpjbf  = (__bf16*)carve((size_t)3 * CC2 * 2);
  __bf16* aowbf   = (__bf16*)carve((size_t)CC2 * 2);
  (void)ws_size; (void)n_in; (void)in_sizes; (void)out_size;

  // ---- weight / feature conversion ----
  auto g4 = [](long n) { return (unsigned)((n + 1023) / 1024); };
  mps_cvt_bf16<<<g4((long)NA * CC), 256, 0, stream>>>(x_a, xa_bf, (long)NA * CC);
  mps_cvt_bf16<<<g4((long)NB * CC), 256, 0, stream>>>(x_b, xb_bf, (long)NB * CC);
  // in_proj_w [768,256] and attn_out_w [256,256] multiply as x@W^T -> already [N][K]
  mps_cvt_bf16<<<g4(3L * CC2), 256, 0, stream>>>(in_proj_w, inpjbf, 3L * CC2);
  mps_cvt_bf16<<<g4((long)CC2), 256, 0, stream>>>(attn_out_w, aowbf, (long)CC2);
  unsigned gt = (unsigned)((CC2 + 255) / 256);
  for (int m = 0; m < MM; ++m) {
    mps_cvt_transpose_bf16<<<gt, 256, 0, stream>>>(wl1 + (size_t)m * CC2, wl1t + (size_t)m * CC2, CC, CC);
    mps_cvt_transpose_bf16<<<gt, 256, 0, stream>>>(wr1 + (size_t)m * CC2, wr1t + (size_t)m * CC2, CC, CC);
    mps_cvt_transpose_bf16<<<gt, 256, 0, stream>>>(wl2 + (size_t)m * CC2, wl2t + (size_t)m * CC2, CC, CC);
    mps_cvt_transpose_bf16<<<gt, 256, 0, stream>>>(wr2 + (size_t)m * CC2, wr2t + (size_t)m * CC2, CC, CC);
    mps_cvt_transpose_bf16<<<gt, 256, 0, stream>>>(outw + (size_t)m * CC2, outwt + (size_t)m * CC2, CC, CC);
  }
  mps_cvt_transpose_bf16<<<gt, 256, 0, stream>>>(mlp1_w, mlp1t, CC, CC);

  const unsigned gsc = (unsigned)(((size_t)EE * 64 + 255) / 256);
  // ---- per-metapath GNN ----
  for (int m = 0; m < MM; ++m) {
    const int* ab_src = edges_ab + (size_t)m * 2 * EE;
    const int* ab_dst = ab_src + EE;
    const int* ba_src = edges_ba + (size_t)m * 2 * EE;
    const int* ba_dst = ba_src + EE;

    // SAGE A->B
    (void)hipMemsetAsync(agg, 0, (size_t)NB * CC * 4, stream);
    (void)hipMemsetAsync(cnt, 0, (size_t)NB * 4, stream);
    mps_scatter_f32<<<gsc, 256, 0, stream>>>(x_a, ab_src, ab_dst, agg, cnt, EE);
    mps_mean_bf16<<<(unsigned)(((size_t)NB * 64 + 255) / 256), 256, 0, stream>>>(agg, cnt, meanbf, NB);
    dim3 g1((NB + 127) / 128, CC / 128);
    mps_wmma_gemm<<<g1, 256, SMEM2, stream>>>(meanbf, wl1t + (size_t)m * CC2,
                                              xb_bf,  wr1t + (size_t)m * CC2,
                                              b1 + (size_t)m * CC, nullptr,
                                              nullptr, h_bf, NB, CC, CC, CC, 1);

    // SAGE B->A
    (void)hipMemsetAsync(agg, 0, (size_t)NA * CC * 4, stream);
    (void)hipMemsetAsync(cnt, 0, (size_t)NA * 4, stream);
    mps_scatter_bf16<<<gsc, 256, 0, stream>>>(h_bf, ba_src, ba_dst, agg, cnt, EE);
    mps_mean_bf16<<<(unsigned)(((size_t)NA * 64 + 255) / 256), 256, 0, stream>>>(agg, cnt, meanbf, NA);
    dim3 g2((NA + 127) / 128, CC / 128);
    mps_wmma_gemm<<<g2, 256, SMEM2, stream>>>(meanbf, wl2t + (size_t)m * CC2,
                                              xa_bf,  wr2t + (size_t)m * CC2,
                                              b2 + (size_t)m * CC, nullptr,
                                              nullptr, h_bf, NA, CC, CC, CC, 1);

    // out_proj, scaled by mp_weights[m], strided into xstack[:, m, :]
    mps_wmma_gemm<<<g2, 256, SMEM1, stream>>>(h_bf, outwt + (size_t)m * CC2,
                                              nullptr, nullptr,
                                              outb + (size_t)m * CC, mp_w + m,
                                              nullptr, xstack + (size_t)m * CC,
                                              NA, CC, CC, MM * CC, 0);
  }

  // ---- attention ----
  dim3 gq((NA * MM + 127) / 128, (3 * CC) / 128);
  mps_wmma_gemm<<<gq, 256, SMEM1, stream>>>(xstack, inpjbf, nullptr, nullptr,
                                            in_proj_b, nullptr, nullptr, qkv,
                                            NA * MM, 3 * CC, CC, 3 * CC, 0);
  mps_attention<<<NA, 128, 0, stream>>>(qkv, xstack /* o aliases xstack */, NA);
  dim3 ga((NA * MM + 127) / 128, CC / 128);
  mps_wmma_gemm<<<ga, 256, SMEM1, stream>>>(xstack, aowbf, nullptr, nullptr,
                                            attn_out_b, nullptr, attn_f, nullptr,
                                            NA * MM, CC, CC, CC, 0);
  mps_pool<<<(unsigned)(((size_t)NA * CC + 255) / 256), 256, 0, stream>>>(attn_f, pooled, NA);

  // ---- MLP head ----
  dim3 gm((NA + 127) / 128, CC / 128);
  mps_wmma_gemm<<<gm, 256, SMEM1, stream>>>(pooled, mlp1t, nullptr, nullptr,
                                            mlp1_b, nullptr, nullptr, hmlp,
                                            NA, CC, CC, CC, 1);
  mps_final<<<(NA + 7) / 8, 256, 0, stream>>>(hmlp, mlp2_w, mlp2_b, out, NA);
}
